// lstm_generator_14156212208103
// MI455X (gfx1250) — matrix operand order repaired and checked by reading, not compiled, not run on hardware
//
#include <hip/hip_runtime.h>

// ---------------------------------------------------------------------------
// 2-layer LSTM (B=64, T=2048, IN=128, H=256) + output projection (OUT=64)
// Kernel A: persistent recurrent kernel, weights LDS-resident, f16 WMMA,
//           h-state exchanged via f16 mirror + global_load_async_to_lds_b128.
// Kernel B: output projection GEMM, f16 WMMA.
// ---------------------------------------------------------------------------

typedef __attribute__((ext_vector_type(16))) _Float16 v16h;
typedef __attribute__((ext_vector_type(8)))  float    v8f;

#define B_    64
#define T_    2048
#define IN_   128
#define H_    256
#define OUT_  64

#define NWG     32          // persistent workgroups (co-resident, << occupancy limit)
#define HS      8           // hidden units owned per WG (per layer)
#define NS      (4 * HS)    // gate-slice width = 32 (i,f,g,o x HS)
#define THREADS 256         // 8 wave32

// ---------------- fast elementwise ----------------
__device__ inline float sigmoid_(float x) { return 1.0f / (1.0f + __expf(-x)); }
__device__ inline float tanh_(float x) {
  float e = __expf(-2.0f * x);
  return (1.0f - e) / (1.0f + e);
}

// ---------------- CDNA5 async copy: global -> LDS (ASYNCcnt-tracked DMA) ----
__device__ inline void async_copy_b128(unsigned lds_addr, const void* gptr) {
  asm volatile("global_load_async_to_lds_b128 %0, %1, off"
               :: "v"(lds_addr), "v"(gptr)
               : "memory");
}
__device__ inline void wait_async0() {
  asm volatile("s_wait_asynccnt 0" ::: "memory");
}

// ---------------- WMMA fragment helpers (ISA 7.12.2 layouts) ----------------
// A (16x32 f16): lanes 0-15 -> M=lane, K = {0..7,16..23}; lanes 16-31 -> K += 8
__device__ inline v16h load_a_lds(const _Float16* A, int ld, int m0, int k0, int lane) {
  const _Float16* p = A + (m0 + (lane & 15)) * ld + k0 + ((lane >> 4) << 3);
  v16h a;
#pragma unroll
  for (int e = 0; e < 8; ++e) { a[e] = p[e]; a[8 + e] = p[16 + e]; }
  return a;
}

// A fragment gathered from a global fp32 row-major matrix (cvt f32->f16 on the fly)
__device__ inline v16h load_a_glb(const float* base, long ld, long m0, int k0, int lane) {
  const float* p = base + (m0 + (long)(lane & 15)) * ld + k0 + ((lane >> 4) << 3);
  v16h a;
#pragma unroll
  for (int e = 0; e < 8; ++e) { a[e] = (_Float16)p[e]; a[8 + e] = (_Float16)p[16 + e]; }
  return a;
}

// B (32x16 f16), B[k][n] = W[n][k]: lane&15 -> N, K = {0..7,16..23}; lanes 16-31 -> K += 8
__device__ inline v16h load_b_lds(const _Float16* W, int ld, int n0, int k0, int lane) {
  const _Float16* p = W + (n0 + (lane & 15)) * ld + k0 + ((lane >> 4) << 3);
  v16h b;
#pragma unroll
  for (int e = 0; e < 8; ++e) { b[e] = p[e]; b[8 + e] = p[16 + e]; }
  return b;
}

// C/D (16x16 f32): VGPR r -> M = r + 8*(lane>>4), N = lane&15
__device__ inline void store_d_lds(float* G, int ldn, int m0, int n0, int lane, v8f c) {
  int n  = n0 + (lane & 15);
  int mb = m0 + ((lane >> 4) << 3);
#pragma unroll
  for (int r = 0; r < 8; ++r) G[(mb + r) * ldn + n] = c[r];
}

__device__ inline v8f wmma_f16(v16h a, v16h b, v8f c) {
  return __builtin_amdgcn_wmma_f32_16x16x32_f16(false, a, false, b, (short)0, c,
                                                false, false);
}

// ---------------- device-scope barrier across the 32 persistent WGs --------
__device__ inline void grid_barrier(unsigned* bar, unsigned* epoch) {
  __threadfence();            // release our h/c/hx stores to device scope
  __syncthreads();
  if (threadIdx.x == 0) {
    ++(*epoch);
    __hip_atomic_fetch_add(bar, 1u, __ATOMIC_RELEASE, __HIP_MEMORY_SCOPE_AGENT);
    unsigned tgt = (*epoch) * NWG;
    while (__hip_atomic_load(bar, __ATOMIC_ACQUIRE, __HIP_MEMORY_SCOPE_AGENT) < tgt) {
      __builtin_amdgcn_s_sleep(2);
    }
  }
  __syncthreads();
  __threadfence();            // acquire: invalidate WGP$ before reading peers' state
}

// ---------------------------------------------------------------------------
// Kernel A: persistent recurrence
// ---------------------------------------------------------------------------
__global__ __launch_bounds__(THREADS) void lstm_recurrent_kernel(
    const float* __restrict__ input,
    const float* __restrict__ W_ih0, const float* __restrict__ W_hh0,
    const float* __restrict__ b_ih0, const float* __restrict__ b_hh0,
    const float* __restrict__ W_ih1, const float* __restrict__ W_hh1,
    const float* __restrict__ b_ih1, const float* __restrict__ b_hh1,
    float* __restrict__ h_out,   // [2][B][T][H]
    float* __restrict__ c_out,   // [2][B][T][H]
    _Float16* __restrict__ hx0,  // f16 mirror of h0 state [B][H] (in d_ws)
    _Float16* __restrict__ hx1,  // f16 mirror of h1 state [B][H] (in d_ws)
    unsigned* __restrict__ bar) {
  // LDS: ~128.3 KB
  __shared__ _Float16 sWih0[NS * IN_];   //  8 KB  (gate-slice rows of W_ih0)
  __shared__ _Float16 sWhh0[NS * H_];    // 16 KB
  __shared__ _Float16 sWih1[NS * H_];    // 16 KB
  __shared__ _Float16 sWhh1[NS * H_];    // 16 KB
  __shared__ _Float16 sH0[B_ * H_];      // 32 KB  full h0 state (f16)
  __shared__ _Float16 sH1[B_ * H_];      // 32 KB  full h1 state (f16)
  __shared__ float    sG[B_ * NS];       //  8 KB  gate pre-activations (f32)
  __shared__ float    sB0[NS], sB1[NS];

  const int tid  = threadIdx.x;
  const int wg   = blockIdx.x;
  const int lane = tid & 31;
  const int wave = tid >> 5;

  // ---- load this WG's weight slices once (fp32 -> f16), resident for all T ----
  for (int f = tid; f < NS * IN_; f += THREADS) {
    int r = f / IN_, k = f % IN_;
    int g = r / HS, u = r % HS;
    sWih0[f] = (_Float16)W_ih0[(g * H_ + wg * HS + u) * IN_ + k];
  }
  for (int f = tid; f < NS * H_; f += THREADS) {
    int r = f >> 8, k = f & (H_ - 1);
    int g = r / HS, u = r % HS;
    int R = g * H_ + wg * HS + u;
    sWhh0[f] = (_Float16)W_hh0[R * H_ + k];
    sWih1[f] = (_Float16)W_ih1[R * H_ + k];
    sWhh1[f] = (_Float16)W_hh1[R * H_ + k];
  }
  if (tid < NS) {
    int g = tid / HS, u = tid % HS;
    int R = g * H_ + wg * HS + u;
    sB0[tid] = b_ih0[R] + b_hh0[R];
    sB1[tid] = b_ih1[R] + b_hh1[R];
  }
  for (int f = tid; f < B_ * H_; f += THREADS) {
    sH0[f] = (_Float16)0.0f;
    sH1[f] = (_Float16)0.0f;
  }
  __syncthreads();

  // cell state lives in registers for the whole sequence
  float c0r[2] = {0.0f, 0.0f};
  float c1r[2] = {0.0f, 0.0f};
  unsigned epoch = 0;

  // tile ownership: 4 M-tiles (batch) x 2 N-tiles (gate slice) = 8 waves
  const int mtile = (wave & 3) << 4;
  const int ntile = (wave >> 2) << 4;

  const unsigned lds_h0 = (unsigned)(uintptr_t)&sH0[0];  // LDS byte address
  const unsigned lds_h1 = (unsigned)(uintptr_t)&sH1[0];

  for (int t = 0; t < T_; ++t) {
    // ================= layer 0: g0 = x(t)@Wih0s^T + h0(t-1)@Whh0s^T =========
    v8f acc = {};
    const float* xbase = input + (long)t * IN_;   // row stride T*IN between batches
#pragma unroll
    for (int k0 = 0; k0 < IN_; k0 += 32) {
      v16h a = load_a_glb(xbase, (long)T_ * IN_, (long)mtile, k0, lane);
      v16h b = load_b_lds(sWih0, IN_, ntile, k0, lane);
      acc = wmma_f16(a, b, acc);
    }
#pragma unroll
    for (int k0 = 0; k0 < H_; k0 += 32) {
      v16h a = load_a_lds(sH0, H_, mtile, k0, lane);
      v16h b = load_b_lds(sWhh0, H_, ntile, k0, lane);
      acc = wmma_f16(a, b, acc);
    }
    store_d_lds(sG, NS, mtile, ntile, lane, acc);
    __syncthreads();

    // elementwise layer 0; publish fp32 outputs + f16 exchange mirror
#pragma unroll
    for (int i = 0; i < 2; ++i) {
      int p  = i * THREADS + tid;        // 512 (batch, unit) pairs
      int bb = p >> 3, u = p & (HS - 1);
      float ig = sG[bb * NS + 0 * HS + u] + sB0[0 * HS + u];
      float fg = sG[bb * NS + 1 * HS + u] + sB0[1 * HS + u];
      float gg = sG[bb * NS + 2 * HS + u] + sB0[2 * HS + u];
      float og = sG[bb * NS + 3 * HS + u] + sB0[3 * HS + u];
      float c  = sigmoid_(fg) * c0r[i] + sigmoid_(ig) * tanh_(gg);
      float h  = sigmoid_(og) * tanh_(c);
      c0r[i]   = c;
      long idx = ((long)bb * T_ + t) * H_ + wg * HS + u;   // layer 0 block
      h_out[idx] = h;
      c_out[idx] = c;
      hx0[bb * H_ + wg * HS + u] = (_Float16)h;            // f16 mirror
    }

    // prefetch next timestep's input rows while we wait at the barrier
    if (t + 1 < T_)
      __builtin_prefetch(input + ((long)(tid & (B_ - 1)) * T_ + (t + 1)) * IN_, 0, 1);

    grid_barrier(bar, &epoch);   // publish h0(t)

    // DMA the freshly published f16 states straight into LDS (no VGPR round-trip)
    {
      const char* g0 = (const char*)hx0;
#pragma unroll
      for (int i = 0; i < (B_ * H_ * 2) / (THREADS * 16); ++i) {   // 8 x b128/thread
        int q = (i * THREADS + tid) * 16;
        async_copy_b128(lds_h0 + q, g0 + q);
      }
      if (t > 0) {
        const char* g1 = (const char*)hx1;
#pragma unroll
        for (int i = 0; i < (B_ * H_ * 2) / (THREADS * 16); ++i) {
          int q = (i * THREADS + tid) * 16;
          async_copy_b128(lds_h1 + q, g1 + q);
        }
      }
      wait_async0();
    }
    __syncthreads();

    // ================= layer 1: g1 = h0(t)@Wih1s^T + h1(t-1)@Whh1s^T ========
    acc = (v8f){};
#pragma unroll
    for (int k0 = 0; k0 < H_; k0 += 32) {
      v16h a = load_a_lds(sH0, H_, mtile, k0, lane);
      v16h b = load_b_lds(sWih1, H_, ntile, k0, lane);
      acc = wmma_f16(a, b, acc);
    }
#pragma unroll
    for (int k0 = 0; k0 < H_; k0 += 32) {
      v16h a = load_a_lds(sH1, H_, mtile, k0, lane);
      v16h b = load_b_lds(sWhh1, H_, ntile, k0, lane);
      acc = wmma_f16(a, b, acc);
    }
    store_d_lds(sG, NS, mtile, ntile, lane, acc);
    __syncthreads();

#pragma unroll
    for (int i = 0; i < 2; ++i) {
      int p  = i * THREADS + tid;
      int bb = p >> 3, u = p & (HS - 1);
      float ig = sG[bb * NS + 0 * HS + u] + sB1[0 * HS + u];
      float fg = sG[bb * NS + 1 * HS + u] + sB1[1 * HS + u];
      float gg = sG[bb * NS + 2 * HS + u] + sB1[2 * HS + u];
      float og = sG[bb * NS + 3 * HS + u] + sB1[3 * HS + u];
      float c  = sigmoid_(fg) * c1r[i] + sigmoid_(ig) * tanh_(gg);
      float h  = sigmoid_(og) * tanh_(c);
      c1r[i]   = c;
      long idx = ((long)(B_ + bb) * T_ + t) * H_ + wg * HS + u;   // layer 1 block
      h_out[idx] = h;
      c_out[idx] = c;
      hx1[bb * H_ + wg * HS + u] = (_Float16)h;            // f16 mirror
    }

    grid_barrier(bar, &epoch);   // publish h1(t)
  }
}

// ---------------------------------------------------------------------------
// Kernel B: y[B*T, OUT] = h1[B*T, H] @ W_out^T + b_out   (WMMA GEMM)
// ---------------------------------------------------------------------------
__global__ __launch_bounds__(256) void lstm_outproj_kernel(
    const float* __restrict__ h1,     // hidden_features layer-1 base: [B*T][H]
    const float* __restrict__ W_out,  // [OUT][H]
    const float* __restrict__ b_out,  // [OUT]
    float* __restrict__ y) {          // [B*T][OUT]
  __shared__ _Float16 sW[OUT_ * H_];  // 32 KB
  __shared__ float    sBias[OUT_];

  const int tid  = threadIdx.x;
  const int lane = tid & 31;
  const int wave = tid >> 5;

  for (int f = tid; f < OUT_ * H_; f += 256) sW[f] = (_Float16)W_out[f];
  if (tid < OUT_) sBias[tid] = b_out[tid];
  __syncthreads();

  const long r0 = (long)blockIdx.x * 128 + (long)wave * 16;  // 16 rows per wave

  v8f acc[4];
#pragma unroll
  for (int j = 0; j < 4; ++j) acc[j] = (v8f){};

#pragma unroll
  for (int k0 = 0; k0 < H_; k0 += 32) {
    v16h a = load_a_glb(h1, (long)H_, r0, k0, lane);  // reuse A across 4 N-tiles
#pragma unroll
    for (int j = 0; j < 4; ++j) {
      v16h b = load_b_lds(sW, H_, j * 16, k0, lane);
      acc[j] = wmma_f16(a, b, acc[j]);
    }
  }

#pragma unroll
  for (int j = 0; j < 4; ++j) {
    int n = j * 16 + (lane & 15);
    float bias = sBias[n];
#pragma unroll
    for (int r = 0; r < 8; ++r) {
      long m = r + ((lane >> 4) << 3);
      y[(r0 + m) * OUT_ + n] = acc[j][r] + bias;
    }
  }
}

// ---------------------------------------------------------------------------
extern "C" void kernel_launch(void* const* d_in, const int* in_sizes, int n_in,
                              void* d_out, int out_size, void* d_ws, size_t ws_size,
                              hipStream_t stream) {
  const float* input = (const float*)d_in[0];
  const float* W_ih0 = (const float*)d_in[1];
  const float* W_hh0 = (const float*)d_in[2];
  const float* b_ih0 = (const float*)d_in[3];
  const float* b_hh0 = (const float*)d_in[4];
  const float* W_ih1 = (const float*)d_in[5];
  const float* W_hh1 = (const float*)d_in[6];
  const float* b_ih1 = (const float*)d_in[7];
  const float* b_hh1 = (const float*)d_in[8];
  const float* W_out = (const float*)d_in[9];
  const float* b_out = (const float*)d_in[10];

  float* y_out = (float*)d_out;                          // [B][T][OUT]
  float* h_out = y_out + (size_t)B_ * T_ * OUT_;         // [2][B][T][H]
  float* c_out = h_out + (size_t)2 * B_ * T_ * H_;       // [2][B][T][H]

  // d_ws layout: [0,256) barrier counter; then two 32 KB f16 h-state mirrors
  char* ws = (char*)d_ws;
  unsigned*  bar = (unsigned*)ws;
  _Float16*  hx0 = (_Float16*)(ws + 256);
  _Float16*  hx1 = (_Float16*)(ws + 256 + (size_t)B_ * H_ * sizeof(_Float16));

  // zero the inter-WG barrier counter (d_ws is poisoned by the harness)
  hipMemsetAsync(d_ws, 0, 256, stream);

  hipLaunchKernelGGL(lstm_recurrent_kernel, dim3(NWG), dim3(THREADS), 0, stream,
                     input, W_ih0, W_hh0, b_ih0, b_hh0, W_ih1, W_hh1, b_ih1, b_hh1,
                     h_out, c_out, hx0, hx1, bar);

  const float* h1_base = h_out + (size_t)B_ * T_ * H_;   // layer-1 hidden block
  hipLaunchKernelGGL(lstm_outproj_kernel, dim3((B_ * T_) / 128), dim3(256), 0, stream,
                     h1_base, W_out, b_out, y_out);
}
